// BahdanauAttention_67121748902264
// MI455X (gfx1250) — matrix operand order repaired and checked by reading, not compiled, not run on hardware
//
#include <hip/hip_runtime.h>
#include <hip/hip_bf16.h>

typedef __attribute__((ext_vector_type(16))) __bf16 v16bf;
typedef __attribute__((ext_vector_type(8)))  float  v8f;

#define S_LEN 2048
#define B_DIM 32
#define D_DIM 1024
#define E_DIM 1024
#define R_ROWS (S_LEN * B_DIM)   /* 65536 flattened (s,b) rows */
#define M_TILE 32                /* rows per block (2 row-tiles of 16) */
#define A_PITCH 1032             /* bf16 elems per LDS row: 1024 + 8 -> rows rotate banks */

// Branchless tanh: prefer the hardware trans op if the builtin exists.
__device__ __forceinline__ float fast_tanh(float x) {
#if __has_builtin(__builtin_amdgcn_tanhf)
    return __builtin_amdgcn_tanhf(x);
#elif __has_builtin(__builtin_amdgcn_tanh_f32)
    return __builtin_amdgcn_tanh_f32(x);
#else
    float ax = fabsf(x);
    float e  = __expf(2.0f * ax);                 // overflow -> inf -> t = 1
    float t  = 1.0f - 2.0f / (e + 1.0f);
    return copysignf(t, x);
#endif
}

// ---------------------------------------------------------------- zero output
__global__ void k_zero(float* __restrict__ out) {
    out[blockIdx.x * 256 + threadIdx.x] = 0.0f;
}

// ----------------------------------------------------- Wc fp32 -> bf16 (2 MB)
__global__ void k_cvt_bf16(const float* __restrict__ src, __bf16* __restrict__ dst) {
    int i = (blockIdx.x * 256 + threadIdx.x) * 4;
    float4 f = *(const float4*)(src + i);
    dst[i + 0] = (__bf16)f.x;
    dst[i + 1] = (__bf16)f.y;
    dst[i + 2] = (__bf16)f.z;
    dst[i + 3] = (__bf16)f.w;
}

// ------------------------------------- weighted_s[b,e] = dec[b,:] . Wb[e,:]
__global__ void k_ws(const float* __restrict__ dec, const float* __restrict__ Wb,
                     float* __restrict__ wsm) {
    int idx = blockIdx.x * 256 + threadIdx.x;   // 32768 outputs
    int b = idx & 31;
    int e = idx >> 5;                           // wave shares one Wb row (broadcast)
    const float4* dp = (const float4*)(dec + (size_t)b * D_DIM);
    const float4* wp = (const float4*)(Wb  + (size_t)e * D_DIM);
    float acc = 0.0f;
    #pragma unroll 4
    for (int j = 0; j < D_DIM / 4; ++j) {
        float4 a = dp[j], w = wp[j];
        acc += a.x * w.x + a.y * w.y + a.z * w.z + a.w * w.w;
    }
    wsm[(size_t)b * D_DIM + e] = acc;
}

// --------------------------------------------------------------- fused score
// score[r] = sum_d tanh( (enc_row[r] . Wc[d,:]) + ws[b(r),d] ) * Wa[d]
// Block: 32 rows (2 row-tiles). 8 waves x 128 d-cols, register-blocked 2x4:
// per K-step 2 A frags (LDS) + 4 B frags (L2) feed 8 WMMAs.
__global__ __launch_bounds__(256) void k_score(
        const float*  __restrict__ enc,
        const __bf16* __restrict__ wc,     // bf16 Wc, row-major [d][e] (L2 resident)
        const float*  __restrict__ wsm,    // weighted_s [b][d]
        const float*  __restrict__ Wa,
        float*        __restrict__ score)  // [R] = [s*B + b]
{
    __shared__ __bf16 Albs[M_TILE * A_PITCH];   // 66 KB
    __shared__ float  score_lds[M_TILE];

    const int t = threadIdx.x;
    const int rowbase = blockIdx.x * M_TILE;

    // Stage A tile: 32 rows x 1024 e, fp32 -> bf16 into LDS
    {
        const float4* src = (const float4*)(enc + (size_t)rowbase * E_DIM);
        #pragma unroll
        for (int j = 0; j < M_TILE; ++j) {
            int idx4 = t + j * 256;            // 0..8191 float4s
            int row  = idx4 >> 8;
            int col4 = idx4 & 255;
            float4 f = src[idx4];
            __bf16* dst = &Albs[row * A_PITCH + col4 * 4];
            dst[0] = (__bf16)f.x; dst[1] = (__bf16)f.y;
            dst[2] = (__bf16)f.z; dst[3] = (__bf16)f.w;
        }
    }
    if (t < M_TILE) score_lds[t] = 0.0f;
    __syncthreads();

    const int wave = t >> 5;
    const int lane = t & 31;
    const int hf   = lane >> 4;      // half-wave select
    const int n    = lane & 15;      // A frag row / B frag column / C col

    float p0[8] = {0.f,0.f,0.f,0.f,0.f,0.f,0.f,0.f};
    float p1[8] = {0.f,0.f,0.f,0.f,0.f,0.f,0.f,0.f};

    union F { v16bf v; uint4 q[2]; };

    const __bf16* arow0 = &Albs[n * A_PITCH + 8 * hf];           // tile0 row n
    const __bf16* arow1 = &Albs[(16 + n) * A_PITCH + 8 * hf];    // tile1 row n

    #pragma unroll 1
    for (int g = 0; g < 2; ++g) {                // two groups of 4 d-tiles
        // Memory clobber: stops LICM from hoisting the (kk-only-dependent)
        // LDS fragment loads across group iterations -> no register blowup/spill.
        asm volatile("" ::: "memory");

        const int dbase = wave * 128 + g * 64;
        const __bf16* brow = wc + (size_t)(dbase + n) * E_DIM + 8 * hf;

        v8f acc0[4] = {v8f{}, v8f{}, v8f{}, v8f{}};
        v8f acc1[4] = {v8f{}, v8f{}, v8f{}, v8f{}};

        #pragma unroll 2
        for (int kk = 0; kk < 32; ++kk) {
            F a0, a1;
            {
                const __bf16* p = arow0 + kk * 32;
                a0.q[0] = *(const uint4*)(p);
                a0.q[1] = *(const uint4*)(p + 16);
                const __bf16* q = arow1 + kk * 32;
                a1.q[0] = *(const uint4*)(q);
                a1.q[1] = *(const uint4*)(q + 16);
            }
            F bf[4];
            #pragma unroll
            for (int j = 0; j < 4; ++j) {
                const __bf16* p = brow + (size_t)j * 16 * E_DIM + kk * 32;
                bf[j].q[0] = *(const uint4*)(p);
                bf[j].q[1] = *(const uint4*)(p + 16);
            }
            #pragma unroll
            for (int j = 0; j < 4; ++j) {
                acc0[j] = __builtin_amdgcn_wmma_f32_16x16x32_bf16(
                            false, a0.v, false, bf[j].v, (short)0, acc0[j], false, false);
                acc1[j] = __builtin_amdgcn_wmma_f32_16x16x32_bf16(
                            false, a1.v, false, bf[j].v, (short)0, acc1[j], false, false);
            }
        }

        // Epilogue for this group's 4 d-tiles: tanh + Wa partial dot, in regs
        #pragma unroll
        for (int j = 0; j < 4; ++j) {
            const int d  = dbase + j * 16 + n;
            const float wa = Wa[d];
            #pragma unroll
            for (int i = 0; i < 8; ++i) {
                const int m0 = i + 8 * hf;               // C layout: VGPR i -> M=i / M=i+8
                const int b0 = (rowbase + m0) & 31;      // r = s*B + b -> b = r & 31
                p0[i] += fast_tanh(acc0[j][i] + wsm[(size_t)b0 * D_DIM + d]) * wa;
                const int m1 = 16 + m0;
                const int b1 = (rowbase + m1) & 31;
                p1[i] += fast_tanh(acc1[j][i] + wsm[(size_t)b1 * D_DIM + d]) * wa;
            }
        }
    }

    // Reduce row partials across 16-lane groups, combine waves via LDS atomics
    #pragma unroll
    for (int i = 0; i < 8; ++i) {
        float v = p0[i];
        v += __shfl_xor(v, 1, 32);
        v += __shfl_xor(v, 2, 32);
        v += __shfl_xor(v, 4, 32);
        v += __shfl_xor(v, 8, 32);
        if (n == 0) atomicAdd(&score_lds[i + 8 * hf], v);
        float w = p1[i];
        w += __shfl_xor(w, 1, 32);
        w += __shfl_xor(w, 2, 32);
        w += __shfl_xor(w, 4, 32);
        w += __shfl_xor(w, 8, 32);
        if (n == 0) atomicAdd(&score_lds[16 + i + 8 * hf], w);
    }
    __syncthreads();
    if (t < M_TILE) score[rowbase + t] = score_lds[t];
}

// --------------------------------------------- softmax over s for each b col
__global__ void k_softmax(const float* __restrict__ score, float* __restrict__ attn) {
    __shared__ float red[256];
    const int b = blockIdx.x, t = threadIdx.x;

    float m = -1e30f;
    for (int s = t; s < S_LEN; s += 256) m = fmaxf(m, score[s * B_DIM + b]);
    red[t] = m; __syncthreads();
    for (int o = 128; o > 0; o >>= 1) { if (t < o) red[t] = fmaxf(red[t], red[t + o]); __syncthreads(); }
    const float gmax = red[0];
    __syncthreads();

    float sum = 0.0f;
    for (int s = t; s < S_LEN; s += 256) sum += expf(score[s * B_DIM + b] - gmax);
    red[t] = sum; __syncthreads();
    for (int o = 128; o > 0; o >>= 1) { if (t < o) red[t] += red[t + o]; __syncthreads(); }
    const float inv = 1.0f / red[0];

    for (int s = t; s < S_LEN; s += 256)
        attn[s * B_DIM + b] = expf(score[s * B_DIM + b] - gmax) * inv;
}

// ------------------------- context[b,e] = sum_s attn[s,b] * enc[s,b,e]
__global__ void k_context(const float* __restrict__ attn, const float* __restrict__ enc,
                          float* __restrict__ out) {
    const int e  = blockIdx.x * 256 + threadIdx.x;  // 4 chunks of 256
    const int b  = blockIdx.y;                       // 32
    const int s0 = blockIdx.z * 256;                 // 8 S-chunks
    float acc = 0.0f;
    for (int s = s0; s < s0 + 256; ++s)
        acc += attn[s * B_DIM + b] * enc[((size_t)s * B_DIM + b) * E_DIM + e];
    atomicAdd(&out[b * E_DIM + e], acc);
}

// ---------------------------------------------------------------------- entry
extern "C" void kernel_launch(void* const* d_in, const int* in_sizes, int n_in,
                              void* d_out, int out_size, void* d_ws, size_t ws_size,
                              hipStream_t stream) {
    const float* dec = (const float*)d_in[0];   // [B, D]
    const float* enc = (const float*)d_in[1];   // [S, B, E2]
    const float* Wb  = (const float*)d_in[2];   // [D, D]
    const float* Wc  = (const float*)d_in[3];   // [D, E2]
    const float* Wa  = (const float*)d_in[4];   // [D]
    float* out = (float*)d_out;                 // [1, B, E2]

    char* ws = (char*)d_ws;
    __bf16* wc_bf = (__bf16*)ws;                              // 2 MB
    float*  score = (float*)(ws + (2u << 20));                // 256 KB
    float*  attn  = (float*)(ws + (2u << 20) + (256u << 10)); // 256 KB
    float*  wsm   = (float*)(ws + (2u << 20) + (512u << 10)); // 128 KB

    k_zero    <<<128,  256, 0, stream>>>(out);
    k_cvt_bf16<<<1024, 256, 0, stream>>>(Wc, wc_bf);
    k_ws      <<<128,  256, 0, stream>>>(dec, Wb, wsm);
    k_score   <<<R_ROWS / M_TILE, 256, 0, stream>>>(enc, wc_bf, wsm, Wa, score);
    k_softmax <<<B_DIM, 256, 0, stream>>>(score, attn);
    k_context <<<dim3(4, B_DIM, 8), 256, 0, stream>>>(attn, enc, out);
}
